// TransNetSweepingExplRhs_48017734370001
// MI455X (gfx1250) — matrix operand order repaired and checked by reading, not compiled, not run on hardware
//
#include <hip/hip_runtime.h>
#include <hip/hip_bf16.h>

// ---------------------------------------------------------------------------
// TransNet sweeping solver, collapsed to GEMMs (see analysis):
//   h = x@W_in^T + b_in ; th = tanh(h); a1 = h + DT*b1
//   S_k = c*I + DT^2*Wk^T Wk  (c = 1 + DT/EPS = 11), solved by Neumann series
//   q1 : S1 q1 = 11*th + DT*(a1@W1)          (3 Neumann iterations, pure GEMM)
//   a2 = a1 - DT*(q1@W1^T) + DT*b2
//   q2 : S2 q2 = 10*th + q1 + DT*(a2@W2)
//   p2 = a2 - DT*(q2@W2^T)
//   out = softmax(p2@W_out^T + b_out)
// GEMMs run on v_wmma_f32_16x16x32_f16 with split-f16 (hi+lo) operands:
// acc += Ahi*Bhi + Ahi*Blo + Alo*Bhi  -> ~fp32 accuracy at f16-WMMA rate.
// B panels are staged with global_load_async_to_lds_b128 (ASYNCcnt).
// ---------------------------------------------------------------------------

#define ASYNC_LDS 1

typedef _Float16 v4h  __attribute__((ext_vector_type(4)));
typedef _Float16 v8h  __attribute__((ext_vector_type(8)));
typedef _Float16 v16h __attribute__((ext_vector_type(16)));
typedef float    v8f  __attribute__((ext_vector_type(8)));

#define K_DIM  1024
#define B_ROWS 2048
#define U_DIM  1024
#define DOUT   1000
#define BM     128
#define BN     64
#define SA     40      // LDS row stride in halfwords (16B-aligned, bank-skewed)
#define DT_C   0.1f
#define C_C    11.0f   // 1 + DT/EPS

__device__ __forceinline__ v8f wmma_f16(v16h a, v16h b, v8f c) {
  // 8 args: (neg_a, A, neg_b, B, c_mod, C, reuse_a, reuse_b)
  return __builtin_amdgcn_wmma_f32_16x16x32_f16(false, a, false, b, (short)0, c,
                                                false, false);
}

// A fragment (16x32 f16, ISA 7.12.2): lane L, M=L&15, half=L>>4;
// f16 elements e=0..7 -> K=8*half+e ; e=8..15 -> K=16+8*half+(e-8).
__device__ __forceinline__ v16h frag_a(const _Float16* rowp, int half) {
  v8h lo = *(const v8h*)(rowp + 8 * half);
  v8h hi = *(const v8h*)(rowp + 16 + 8 * half);
  return __builtin_shufflevector(lo, hi, 0, 1, 2, 3, 4, 5, 6, 7,
                                 8, 9, 10, 11, 12, 13, 14, 15);
}
// B fragment (32x16 f16): lane L, N=L&15, half=L>>4; e=0..7 -> K=8*half+e ; e=8..15 -> K=16+8*half+(e-8).
__device__ __forceinline__ v16h frag_b(const _Float16* rowp, int half) {
  v8h lo = *(const v8h*)(rowp + 8 * half);
  v8h hi = *(const v8h*)(rowp + 16 + 8 * half);
  return __builtin_shufflevector(lo, hi, 0, 1, 2, 3, 4, 5, 6, 7,
                                 8, 9, 10, 11, 12, 13, 14, 15);
}

// Truncate a generic LDS pointer to its 32-bit wave-relative LDS byte offset
// (ISA 10.2: LDS aperture addresses map to LDS by discarding upper bits).
__device__ __forceinline__ unsigned lds_off_u32(const void* p) {
  return (unsigned)(size_t)p;
}

// C = alpha * (A @ Bimg^T) [+ beta1*aux1 + beta2*aux2 + bvscale*bvec[col]]
// A   : f32 [B_ROWS][K_DIM]
// Bimg: split-f16 hi/lo, laid out [N][K_DIM] (row n = output column n)
__global__ __launch_bounds__(256) void gemm_splitf16(
    const float* __restrict__ A, const _Float16* __restrict__ Bh,
    const _Float16* __restrict__ Bl, float* __restrict__ out, int N,
    float alpha, const float* __restrict__ aux1, float beta1,
    const float* __restrict__ aux2, float beta2,
    const float* __restrict__ bvec, float bvscale) {
  __shared__ _Float16 sAh[BM * SA];
  __shared__ _Float16 sAl[BM * SA];
  __shared__ _Float16 sBh[BN * SA];
  __shared__ _Float16 sBl[BN * SA];

  const int tid  = threadIdx.x;
  const int lane = tid & 31;
  const int wid  = tid >> 5;      // 8 waves
  const int wm   = wid >> 1;      // 0..3  (rows of 32)
  const int wn   = wid & 1;       // 0..1  (cols of 32)
  const int half = lane >> 4;
  const int l15  = lane & 15;
  const int m0   = blockIdx.y * BM;
  const int n0   = blockIdx.x * BN;

  // A staging slots: thread covers rows r0+{0,32,64,96}, cols c4..c4+3
  const int r0 = tid >> 3;
  const int c4 = (tid & 7) * 4;
  // B staging slot: one 8-halfword chunk
  const int bn  = tid >> 2;
  const int bc8 = (tid & 3) * 8;
  const int gn  = n0 + bn;

  v8f acc[2][2] = {};

  for (int k0 = 0; k0 < K_DIM; k0 += 32) {
    __syncthreads();
    // ---- stage B panel via async global->LDS (f16 images, [N][K]) ----
    if (gn < N) {
#if ASYNC_LDS
      const unsigned offh = lds_off_u32(sBh + bn * SA + bc8);
      const unsigned offl = lds_off_u32(sBl + bn * SA + bc8);
      const _Float16* gh = Bh + (size_t)gn * K_DIM + k0 + bc8;
      const _Float16* gl = Bl + (size_t)gn * K_DIM + k0 + bc8;
      asm volatile(
          "global_load_async_to_lds_b128 %0, %1, off\n\t"
          "global_load_async_to_lds_b128 %2, %3, off"
          :
          : "v"(offh), "v"(gh), "v"(offl), "v"(gl)
          : "memory");
#else
      v8h bh = *(const v8h*)(Bh + (size_t)gn * K_DIM + k0 + bc8);
      v8h bl = *(const v8h*)(Bl + (size_t)gn * K_DIM + k0 + bc8);
      *(v8h*)(sBh + bn * SA + bc8) = bh;
      *(v8h*)(sBl + bn * SA + bc8) = bl;
#endif
    } else {
      v8h z = {};
      *(v8h*)(sBh + bn * SA + bc8) = z;
      *(v8h*)(sBl + bn * SA + bc8) = z;
    }
    // ---- stage A panel: f32 -> hi/lo f16 (on-the-fly split) ----
#pragma unroll
    for (int rr4 = 0; rr4 < 4; ++rr4) {
      const int r = r0 + 32 * rr4;
      const float4 v =
          *(const float4*)(A + (size_t)(m0 + r) * K_DIM + k0 + c4);
      const _Float16 h0 = (_Float16)v.x, h1 = (_Float16)v.y;
      const _Float16 h2 = (_Float16)v.z, h3 = (_Float16)v.w;
      v4h vh = {h0, h1, h2, h3};
      v4h vl = {(_Float16)(v.x - (float)h0), (_Float16)(v.y - (float)h1),
                (_Float16)(v.z - (float)h2), (_Float16)(v.w - (float)h3)};
      *(v4h*)(sAh + r * SA + c4) = vh;
      *(v4h*)(sAl + r * SA + c4) = vl;
    }
    if (k0 + 32 < K_DIM) {  // next-panel prefetch -> global_prefetch_b8
      __builtin_prefetch(A + (size_t)(m0 + (tid >> 1)) * K_DIM + k0 + 32, 0, 1);
    }
#if ASYNC_LDS
    asm volatile("s_wait_asynccnt 0x0" ::: "memory");
#endif
    __syncthreads();

    v16h aH[2], aL[2], bH[2], bL[2];
#pragma unroll
    for (int mt = 0; mt < 2; ++mt) {
      const _Float16* rp = sAh + (wm * 32 + mt * 16 + l15) * SA;
      const _Float16* rq = sAl + (wm * 32 + mt * 16 + l15) * SA;
      aH[mt] = frag_a(rp, half);
      aL[mt] = frag_a(rq, half);
    }
#pragma unroll
    for (int nt = 0; nt < 2; ++nt) {
      const _Float16* rp = sBh + (wn * 32 + nt * 16 + l15) * SA;
      const _Float16* rq = sBl + (wn * 32 + nt * 16 + l15) * SA;
      bH[nt] = frag_b(rp, half);
      bL[nt] = frag_b(rq, half);
    }
#pragma unroll
    for (int mt = 0; mt < 2; ++mt)
#pragma unroll
      for (int nt = 0; nt < 2; ++nt) {
        acc[mt][nt] = wmma_f16(aH[mt], bH[nt], acc[mt][nt]);
        acc[mt][nt] = wmma_f16(aH[mt], bL[nt], acc[mt][nt]);
        acc[mt][nt] = wmma_f16(aL[mt], bH[nt], acc[mt][nt]);
      }
  }

  // ---- epilogue: C layout (ISA): VGPR v -> row = v + 8*half, col = l15 ----
#pragma unroll
  for (int mt = 0; mt < 2; ++mt)
#pragma unroll
    for (int nt = 0; nt < 2; ++nt) {
      const int col = n0 + wn * 32 + nt * 16 + l15;
      if (col < N) {
#pragma unroll
        for (int v = 0; v < 8; ++v) {
          const int row = m0 + wm * 32 + mt * 16 + v + 8 * half;
          const size_t o = (size_t)row * N + col;
          float r = alpha * acc[mt][nt][v];
          if (aux1) r += beta1 * aux1[o];
          if (aux2) r += beta2 * aux2[o];
          if (bvec) r += bvscale * bvec[col];
          out[o] = r;
        }
      }
    }
}

// ---- weight pre-split kernels --------------------------------------------
__global__ void prep_split_direct(const float* __restrict__ W,
                                  _Float16* __restrict__ oh,
                                  _Float16* __restrict__ ol, int n) {
  const int i = blockIdx.x * 256 + threadIdx.x;
  if (i < n) {
    const float w = W[i];
    const _Float16 h = (_Float16)w;
    oh[i] = h;
    ol[i] = (_Float16)(w - (float)h);
  }
}

// img[n][k] = W[k][n], W is [1024][1024]
__global__ void prep_split_transpose(const float* __restrict__ W,
                                     _Float16* __restrict__ oh,
                                     _Float16* __restrict__ ol) {
  const int i = blockIdx.x * 256 + threadIdx.x;  // i = n*1024 + k
  const int n = i >> 10, k = i & 1023;
  const float w = W[k * 1024 + n];
  const _Float16 h = (_Float16)w;
  oh[i] = h;
  ol[i] = (_Float16)(w - (float)h);
}

// ---- elementwise ----------------------------------------------------------
__global__ void ew_post_h(const float* __restrict__ h,
                          const float* __restrict__ b1,
                          float* __restrict__ th, float* __restrict__ a1,
                          int n) {
  const int i = blockIdx.x * 256 + threadIdx.x;
  if (i < n) {
    const float v = h[i];
    th[i] = tanhf(v);
    a1[i] = v + DT_C * b1[i & (U_DIM - 1)];
  }
}

__global__ void ew_scale(float* __restrict__ q, const float* __restrict__ rr,
                         float s, int n) {
  const int i = blockIdx.x * 256 + threadIdx.x;
  if (i < n) q[i] = rr[i] * s;
}

__global__ __launch_bounds__(256) void softmax_rows(
    const float* __restrict__ logits, float* __restrict__ out, int n) {
  __shared__ float red[256];
  const int row = blockIdx.x, tid = threadIdx.x;
  const float* Lr = logits + (size_t)row * n;
  float m = -3.4e38f;
  for (int c = tid; c < n; c += 256) m = fmaxf(m, Lr[c]);
  red[tid] = m;
  __syncthreads();
  for (int s = 128; s > 0; s >>= 1) {
    if (tid < s) red[tid] = fmaxf(red[tid], red[tid + s]);
    __syncthreads();
  }
  const float mx = red[0];
  __syncthreads();
  float sum = 0.f;
  for (int c = tid; c < n; c += 256) sum += expf(Lr[c] - mx);
  red[tid] = sum;
  __syncthreads();
  for (int s = 128; s > 0; s >>= 1) {
    if (tid < s) red[tid] += red[tid + s];
    __syncthreads();
  }
  const float inv = 1.0f / red[0];
  for (int c = tid; c < n; c += 256)
    out[(size_t)row * n + c] = expf(Lr[c] - mx) * inv;
}

// ---------------------------------------------------------------------------
extern "C" void kernel_launch(void* const* d_in, const int* in_sizes, int n_in,
                              void* d_out, int out_size, void* d_ws,
                              size_t ws_size, hipStream_t stream) {
  (void)in_sizes; (void)n_in; (void)out_size; (void)ws_size;
  const float* x     = (const float*)d_in[0];
  const float* W_in  = (const float*)d_in[1];
  const float* b_in  = (const float*)d_in[2];
  const float* W1    = (const float*)d_in[3];
  const float* b1    = (const float*)d_in[4];
  const float* W2    = (const float*)d_in[5];
  const float* b2    = (const float*)d_in[6];
  const float* W_out = (const float*)d_in[7];
  const float* b_out = (const float*)d_in[8];
  float* out = (float*)d_out;

  char* ws = (char*)d_ws;
  size_t off = 0;
  auto alloc = [&](size_t bytes) -> void* {
    void* p = ws + off;
    off += (bytes + 255) & ~(size_t)255;
    return p;
  };
  const size_t imgU = (size_t)U_DIM * K_DIM * sizeof(_Float16);   // 2 MB
  const size_t imgO = (size_t)DOUT * K_DIM * sizeof(_Float16);
  const size_t actB = (size_t)B_ROWS * U_DIM * sizeof(float);     // 8 MB

  _Float16* WIh  = (_Float16*)alloc(imgU);
  _Float16* WIl  = (_Float16*)alloc(imgU);
  _Float16* W1dh = (_Float16*)alloc(imgU);
  _Float16* W1dl = (_Float16*)alloc(imgU);
  _Float16* W1th = (_Float16*)alloc(imgU);
  _Float16* W1tl = (_Float16*)alloc(imgU);
  _Float16* W2dh = (_Float16*)alloc(imgU);
  _Float16* W2dl = (_Float16*)alloc(imgU);
  _Float16* W2th = (_Float16*)alloc(imgU);
  _Float16* W2tl = (_Float16*)alloc(imgU);
  _Float16* WOh  = (_Float16*)alloc(imgO);
  _Float16* WOl  = (_Float16*)alloc(imgO);
  float* hbuf   = (float*)alloc(actB);  // h, later Neumann temp t
  float* th     = (float*)alloc(actB);
  float* a1     = (float*)alloc(actB);  // a1 -> a2 -> p2 (safe in-place aux)
  float* rr     = (float*)alloc(actB);  // rr1 -> rr2
  float* q      = (float*)alloc(actB);  // q1 -> q2
  float* logits = (float*)alloc((size_t)B_ROWS * DOUT * sizeof(float));

  const int nUU = U_DIM * K_DIM;
  const int nB  = B_ROWS * U_DIM;
  const dim3 t256(256);
  const dim3 gUU((nUU + 255) / 256), gB((nB + 255) / 256);
  const dim3 gOW((DOUT * K_DIM + 255) / 256);

  // weight pre-split (once per launch; deterministic, inputs untouched)
  prep_split_direct<<<gUU, t256, 0, stream>>>(W_in, WIh, WIl, nUU);
  prep_split_direct<<<gUU, t256, 0, stream>>>(W1, W1dh, W1dl, nUU);
  prep_split_direct<<<gUU, t256, 0, stream>>>(W2, W2dh, W2dl, nUU);
  prep_split_direct<<<gOW, t256, 0, stream>>>(W_out, WOh, WOl, DOUT * K_DIM);
  prep_split_transpose<<<gUU, t256, 0, stream>>>(W1, W1th, W1tl);
  prep_split_transpose<<<gUU, t256, 0, stream>>>(W2, W2th, W2tl);

  auto gemm = [&](const float* A, const _Float16* Bh, const _Float16* Bl,
                  float* o, int N, float alpha, const float* x1, float be1,
                  const float* x2, float be2, const float* bv, float bs) {
    dim3 g((N + BN - 1) / BN, B_ROWS / BM);
    gemm_splitf16<<<g, t256, 0, stream>>>(A, Bh, Bl, o, N, alpha, x1, be1, x2,
                                          be2, bv, bs);
  };

  const float invc = 1.0f / C_C;
  const float neuA = -(DT_C * DT_C) / C_C;

  // h = x@W_in^T + b_in
  gemm(x, WIh, WIl, hbuf, U_DIM, 1.f, nullptr, 0.f, nullptr, 0.f, b_in, 1.f);
  // th = tanh(h); a1 = h + DT*b1
  ew_post_h<<<gB, t256, 0, stream>>>(hbuf, b1, th, a1, nB);

  // rr1 = 11*th + DT*(a1@W1)
  gemm(a1, W1th, W1tl, rr, U_DIM, DT_C, th, C_C, nullptr, 0.f, nullptr, 0.f);
  // Neumann: q = rr/c; q <- (rr - DT^2*(q@W1^T)@W1)/c
  ew_scale<<<gB, t256, 0, stream>>>(q, rr, invc, nB);
  for (int it = 0; it < 3; ++it) {
    gemm(q, W1dh, W1dl, hbuf, U_DIM, 1.f, nullptr, 0.f, nullptr, 0.f, nullptr, 0.f);
    gemm(hbuf, W1th, W1tl, q, U_DIM, neuA, rr, invc, nullptr, 0.f, nullptr, 0.f);
  }
  // a2 = a1 - DT*(q1@W1^T) + DT*b2   (in-place over a1)
  gemm(q, W1dh, W1dl, a1, U_DIM, -DT_C, a1, 1.f, nullptr, 0.f, b2, DT_C);

  // rr2 = 10*th + q1 + DT*(a2@W2)
  gemm(a1, W2th, W2tl, rr, U_DIM, DT_C, th, 10.f, q, 1.f, nullptr, 0.f);
  ew_scale<<<gB, t256, 0, stream>>>(q, rr, invc, nB);
  for (int it = 0; it < 3; ++it) {
    gemm(q, W2dh, W2dl, hbuf, U_DIM, 1.f, nullptr, 0.f, nullptr, 0.f, nullptr, 0.f);
    gemm(hbuf, W2th, W2tl, q, U_DIM, neuA, rr, invc, nullptr, 0.f, nullptr, 0.f);
  }
  // p2 = a2 - DT*(q2@W2^T)           (in-place over a2)
  gemm(q, W2dh, W2dl, a1, U_DIM, -DT_C, a1, 1.f, nullptr, 0.f, nullptr, 0.f);

  // logits = p2@W_out^T + b_out ; softmax rows
  gemm(a1, WOh, WOl, logits, DOUT, 1.f, nullptr, 0.f, nullptr, 0.f, b_out, 1.f);
  softmax_rows<<<dim3(B_ROWS), t256, 0, stream>>>(logits, out, DOUT);
}
